// GNNModel_87943750353193
// MI455X (gfx1250) — matrix operand order repaired and checked by reading, not compiled, not run on hardware
//
#include <hip/hip_runtime.h>

// ---------------- problem constants (match reference) ----------------
#define NN 50000
#define EE 800000
#define FF 128
#define HH 256
#define AA 32
#define GG 50
#define BN_EPS 1e-5f

typedef __attribute__((ext_vector_type(4)))  __bf16 v4bf;
typedef __attribute__((ext_vector_type(8)))  __bf16 v8bf;
typedef __attribute__((ext_vector_type(16))) __bf16 v16bf;
typedef __attribute__((ext_vector_type(8)))  float  v8f;

// =====================================================================
// Tiled GEMM  C[M,Nout] = A[M,K] @ W[K,Nout]  (f32 in, bf16 WMMA, f32 out)
// block = 256 threads = 8 waves; block tile 128(M) x 32(N); K step 64.
// Requires K % 64 == 0, Nout % 32 == 0 (head GEMM padded to satisfy).
// Optional fused epilogue: Outp[r,c] = C[r,c] * dis[r]^2 + bias[c]
// (the GCN self-loop + bias init), enabled when Outp != nullptr.
// =====================================================================
#define GEMM_BM 128
#define GEMM_BN 32
#define GEMM_KB 64
#define GSTR   (GEMM_KB + 8)      // 72 bf16 = 144 B row stride (16B aligned)

__global__ __launch_bounds__(256) void gemm_wmma_bf16(
    const float* __restrict__ Amat, const float* __restrict__ Wmat,
    float* __restrict__ Cmat, float* __restrict__ Outp,
    const float* __restrict__ disv, const float* __restrict__ biasv,
    int M, int K, int Nout)
{
  __shared__ __bf16 sA [GEMM_BM][GSTR];   // [row][k]
  __shared__ __bf16 sBt[GEMM_BN][GSTR];   // transposed: [n][k]

  const int tid  = threadIdx.x;
  const int lane = tid & 31;
  const int wv   = tid >> 5;        // 0..7 : 16-row strip
  const int half = lane >> 4;       // 0/1
  const int mrow = lane & 15;       // 0..15

  const int rowBase = blockIdx.x * GEMM_BM;
  const int colBase = blockIdx.y * GEMM_BN;
  const bool fullTile = (rowBase + GEMM_BM) <= M;   // uniform per block

  v8f acc0 = {0.f, 0.f, 0.f, 0.f, 0.f, 0.f, 0.f, 0.f};
  v8f acc1 = acc0;

  for (int k0 = 0; k0 < K; k0 += GEMM_KB) {
    // ---------- phase 1: issue ALL global loads ----------
    float4 av[8];
    if (fullTile) {
#pragma unroll
      for (int it = 0; it < 8; ++it) {
        int idx = tid + it * 256;          // 0..2047 float4 chunks (128 x 16)
        int r   = idx >> 4;
        int c4  = idx & 15;
        av[it] = *reinterpret_cast<const float4*>(
            Amat + (size_t)(rowBase + r) * K + k0 + c4 * 4);
      }
    } else {
#pragma unroll
      for (int it = 0; it < 8; ++it) {
        int idx = tid + it * 256;
        int r   = idx >> 4;
        int c4  = idx & 15;
        int gr  = rowBase + r;
        av[it] = make_float4(0.f, 0.f, 0.f, 0.f);
        if (gr < M)
          av[it] = *reinterpret_cast<const float4*>(
              Amat + (size_t)gr * K + k0 + c4 * 4);
      }
    }
    float4 bv[2];
#pragma unroll
    for (int it = 0; it < 2; ++it) {
      int idx = tid + it * 256;            // 0..511 float4 chunks (64 x 8)
      int r   = idx >> 3;                  // k row 0..63
      int c4  = idx & 7;                   // float4 within 32 cols
      bv[it] = *reinterpret_cast<const float4*>(
          Wmat + (size_t)(k0 + r) * Nout + colBase + c4 * 4);
    }

    // ---------- phase 2: convert + LDS stores ----------
#pragma unroll
    for (int it = 0; it < 8; ++it) {
      int idx = tid + it * 256;
      int r   = idx >> 4;
      int c4  = idx & 15;
      v4bf pk = { (__bf16)av[it].x, (__bf16)av[it].y,
                  (__bf16)av[it].z, (__bf16)av[it].w };
      *reinterpret_cast<v4bf*>(&sA[r][c4 * 4]) = pk;
    }
#pragma unroll
    for (int it = 0; it < 2; ++it) {
      int idx = tid + it * 256;
      int r   = idx >> 3;
      int c4  = idx & 7;
      sBt[c4 * 4 + 0][r] = (__bf16)bv[it].x;
      sBt[c4 * 4 + 1][r] = (__bf16)bv[it].y;
      sBt[c4 * 4 + 2][r] = (__bf16)bv[it].z;
      sBt[c4 * 4 + 3][r] = (__bf16)bv[it].w;
    }
    __syncthreads();

    // ---------- phase 3: fragments + 4 WMMA ----------
    const v8bf* arow  = reinterpret_cast<const v8bf*>(&sA [wv * 16 + mrow][0]);
    const v8bf* b0row = reinterpret_cast<const v8bf*>(&sBt[mrow][0]);
    const v8bf* b1row = reinterpret_cast<const v8bf*>(&sBt[16 + mrow][0]);

#pragma unroll
    for (int s = 0; s < 2; ++s) {
      v8bf alo = arow[s * 4 + half * 2];
      v8bf ahi = arow[s * 4 + half * 2 + 1];
      v16bf afrag = __builtin_shufflevector(alo, ahi,
          0,1,2,3,4,5,6,7,8,9,10,11,12,13,14,15);

      v8bf bl0 = b0row[s * 4 + half * 2];
      v8bf bh0 = b0row[s * 4 + half * 2 + 1];
      v16bf bfrag0 = __builtin_shufflevector(bl0, bh0,
          0,1,2,3,4,5,6,7,8,9,10,11,12,13,14,15);
      acc0 = __builtin_amdgcn_wmma_f32_16x16x32_bf16(
          false, afrag, false, bfrag0, (short)0, acc0, false, false);

      v8bf bl1 = b1row[s * 4 + half * 2];
      v8bf bh1 = b1row[s * 4 + half * 2 + 1];
      v16bf bfrag1 = __builtin_shufflevector(bl1, bh1,
          0,1,2,3,4,5,6,7,8,9,10,11,12,13,14,15);
      acc1 = __builtin_amdgcn_wmma_f32_16x16x32_bf16(
          false, afrag, false, bfrag1, (short)0, acc1, false, false);
    }
    __syncthreads();
  }

  // ---------- epilogue: D layout vgpr v -> row v + 8*half, col = lane&15 ----------
  float bc0 = 0.f, bc1 = 0.f;
  if (Outp) { bc0 = biasv[colBase + mrow]; bc1 = biasv[colBase + 16 + mrow]; }
#pragma unroll
  for (int v = 0; v < 8; ++v) {
    int r = rowBase + wv * 16 + v + 8 * half;
    if (r < M) {
      size_t o = (size_t)r * Nout + colBase + mrow;
      Cmat[o]      = acc0[v];
      Cmat[o + 16] = acc1[v];
      if (Outp) {
        float dd = disv[r];
        float sn = dd * dd;                 // 1/deg (self-loop norm)
        Outp[o]      = acc0[v] * sn + bc0;
        Outp[o + 16] = acc1[v] * sn + bc1;
      }
    }
  }
}

// =====================================================================
// Graph-normalization helpers
// =====================================================================
__global__ void deg_kernel(const int* __restrict__ ei, float* __restrict__ deg, int E)
{
  int e = blockIdx.x * blockDim.x + threadIdx.x;
  if (e < E) atomicAdd(&deg[ei[E + e]], 1.0f);   // dst = row 1
}

__global__ void dis_kernel(float* __restrict__ deg, int n)
{
  int i = blockIdx.x * blockDim.x + threadIdx.x;
  if (i < n) deg[i] = rsqrtf(deg[i] + 1.0f);     // deg^{-1/2} incl. self loop
}

__global__ void enorm_kernel(const int* __restrict__ ei, const float* __restrict__ dis,
                             float* __restrict__ en, int E)
{
  int e = blockIdx.x * blockDim.x + threadIdx.x;
  if (e < E) en[e] = dis[ei[e]] * dis[ei[E + e]];
}

// one wave per edge: lane covers 8 consecutive features (2 x float4 + 8 atomics)
__global__ __launch_bounds__(256) void scatter_kernel(
    const float* __restrict__ lin, const int* __restrict__ ei,
    const float* __restrict__ en, float* __restrict__ out, int E)
{
  int wid  = (int)(((size_t)blockIdx.x * blockDim.x + threadIdx.x) >> 5);
  int lane = threadIdx.x & 31;
  if (wid >= E) return;
  int s = ei[wid], d = ei[E + wid];
  float w = en[wid];
  const float4* sp = reinterpret_cast<const float4*>(lin + (size_t)s * HH) + lane * 2;
  float4 v0 = sp[0], v1 = sp[1];
  float* dp = out + (size_t)d * HH + lane * 8;
  atomicAdd(dp + 0, v0.x * w); atomicAdd(dp + 1, v0.y * w);
  atomicAdd(dp + 2, v0.z * w); atomicAdd(dp + 3, v0.w * w);
  atomicAdd(dp + 4, v1.x * w); atomicAdd(dp + 5, v1.y * w);
  atomicAdd(dp + 6, v1.z * w); atomicAdd(dp + 7, v1.w * w);
}

// =====================================================================
// Batchnorm stats on relu(h): coalesced row-chunk partial sums (thread=column)
// =====================================================================
__global__ __launch_bounds__(HH) void colpartial_kernel(
    const float* __restrict__ h, float* __restrict__ sums, float* __restrict__ sums2,
    int M, int rowsPerBlock)
{
  int c  = threadIdx.x;                     // column (Hd == 256 == blockDim.x)
  int r0 = blockIdx.x * rowsPerBlock;
  int r1 = r0 + rowsPerBlock; if (r1 > M) r1 = M;
  float s = 0.f, s2 = 0.f;
  for (int r = r0; r < r1; ++r) {
    float v = h[(size_t)r * HH + c];
    v = v > 0.f ? v : 0.f;
    s += v; s2 += v * v;
  }
  atomicAdd(&sums[c], s);
  atomicAdd(&sums2[c], s2);
}

__global__ void colfinalize_kernel(float* __restrict__ sums, float* __restrict__ sums2, int M)
{
  int c = blockIdx.x * blockDim.x + threadIdx.x;
  if (c < HH) {
    float m   = sums[c] / (float)M;
    float var = sums2[c] / (float)M - m * m;
    sums[c]  = m;                    // becomes mu
    sums2[c] = rsqrtf(var + BN_EPS); // becomes rstd
  }
}

__global__ void bn_apply_kernel(float* __restrict__ h, const float* __restrict__ mu,
                                const float* __restrict__ rstd, const float* __restrict__ g,
                                const float* __restrict__ be, int M, int Hd)
{
  size_t i = (size_t)blockIdx.x * blockDim.x + threadIdx.x;
  if (i < (size_t)M * Hd) {
    int c = (int)(i % Hd);
    float v = h[i];
    v = v > 0.f ? v : 0.f;
    h[i] = (v - mu[c]) * rstd[c] * g[c] + be[c];
  }
}

__global__ void relu_kernel(float* __restrict__ h, size_t n)
{
  size_t i = (size_t)blockIdx.x * blockDim.x + threadIdx.x;
  if (i < n) { float v = h[i]; h[i] = v > 0.f ? v : 0.f; }
}

// =====================================================================
// Pooling + head
// =====================================================================
__global__ void pool_count_kernel(const int* __restrict__ batch, float* __restrict__ counts)
{
  int n = blockIdx.x * blockDim.x + threadIdx.x;
  if (n < NN) atomicAdd(&counts[batch[n]], 1.0f);
}

__global__ void pool_sum_kernel(const float* __restrict__ h, const int* __restrict__ batch,
                                float* __restrict__ pooled)
{
  size_t i = (size_t)blockIdx.x * blockDim.x + threadIdx.x;
  if (i < (size_t)NN * HH) {
    int n = (int)(i >> 8);
    int f = (int)(i & (HH - 1));
    atomicAdd(&pooled[(size_t)batch[n] * HH + f], h[i]);
  }
}

__global__ void pool_div_kernel(float* __restrict__ pooled, const float* __restrict__ counts)
{
  int i = blockIdx.x * blockDim.x + threadIdx.x;
  if (i < GG * HH) pooled[i] /= fmaxf(counts[i / HH], 1.0f);
}

// z padded to stride 320 (cols 288..319 stay zero)
#define ZKP 320
__global__ void build_z_kernel(const float* __restrict__ pooled,
                               const float* __restrict__ actions, float* __restrict__ z)
{
  int i = blockIdx.x * blockDim.x + threadIdx.x;
  const int ZK = HH + AA;                  // 288 real columns
  if (i < GG * ZK) {
    int g = i / ZK, c = i % ZK;
    z[(size_t)g * ZKP + c] = (c < HH) ? pooled[(size_t)g * HH + c] : actions[c - HH];
  }
}

__global__ void bias_relu_kernel(float* __restrict__ h, const float* __restrict__ b,
                                 int M, int Hd)
{
  size_t i = (size_t)blockIdx.x * blockDim.x + threadIdx.x;
  if (i < (size_t)M * Hd) {
    int c = (int)(i % Hd);
    float v = h[i] + b[c];
    h[i] = v > 0.f ? v : 0.f;
  }
}

__global__ __launch_bounds__(128) void head_final_kernel(
    const float* __restrict__ z2, const float* __restrict__ fW3,
    const float* __restrict__ fb3, float* __restrict__ out)
{
  int g = blockIdx.x, t = threadIdx.x;     // 128 threads = H/2
  __shared__ float ss[128];
  ss[t] = z2[(size_t)g * 128 + t] * fW3[t];
  __syncthreads();
  for (int o = 64; o > 0; o >>= 1) {
    if (t < o) ss[t] += ss[t + o];
    __syncthreads();
  }
  if (t == 0) out[g] = ss[0] + fb3[0];
}

// =====================================================================
// Orchestration
// =====================================================================
extern "C" void kernel_launch(void* const* d_in, const int* in_sizes, int n_in,
                              void* d_out, int out_size, void* d_ws, size_t ws_size,
                              hipStream_t stream)
{
  const float* x       = (const float*)d_in[0];
  const int*   ei      = (const int*)  d_in[1];
  const int*   batch   = (const int*)  d_in[2];
  const float* actions = (const float*)d_in[3];
  const float* W1 = (const float*)d_in[4];  const float* b1 = (const float*)d_in[5];
  const float* W2 = (const float*)d_in[6];  const float* b2 = (const float*)d_in[7];
  const float* W3 = (const float*)d_in[8];  const float* b3 = (const float*)d_in[9];
  const float* g1 = (const float*)d_in[10]; const float* be1 = (const float*)d_in[11];
  const float* g2 = (const float*)d_in[12]; const float* be2 = (const float*)d_in[13];
  const float* g3 = (const float*)d_in[14]; const float* be3 = (const float*)d_in[15];
  const float* fW1 = (const float*)d_in[16]; const float* fb1 = (const float*)d_in[17];
  const float* fW2 = (const float*)d_in[18]; const float* fb2 = (const float*)d_in[19];
  const float* fW3 = (const float*)d_in[20]; const float* fb3 = (const float*)d_in[21];
  float* out = (float*)d_out;

  // ---- workspace carve-up (floats) ----
  const size_t NHf = (size_t)NN * HH;
  float* ws    = (float*)d_ws;
  float* buf0  = ws;                           // N*H  (GEMM output / "lin")
  float* buf1  = buf0 + NHf;                   // N*H  (aggregate / activation)
  float* dis   = buf1 + NHf;                   // N
  float* enorm = dis + NN;                     // E
  float* mu    = enorm + EE;                   // H   (also partial-sum buf)
  float* rstd  = mu + HH;                      // H   (also partial-sumsq buf)
  float* counts= rstd + HH;                    // G
  float* pooled= counts + GG;                  // G*H
  float* zp    = pooled + (size_t)GG * HH;     // G*320 (padded concat input)
  float* wpad  = zp + (size_t)GG * ZKP;        // 320*256 (padded fW1)
  float* z1    = wpad + (size_t)ZKP * HH;      // G*H
  float* z2    = z1 + (size_t)GG * HH;         // G*(H/2)

  const int T = 256;
  const dim3 gemmBlk(256);
  const dim3 gemmGrdN((NN + GEMM_BM - 1) / GEMM_BM, HH / GEMM_BN);

  // ---- one-time inits ----
  hipMemsetAsync(dis, 0, NN * sizeof(float), stream);
  hipMemsetAsync(counts, 0, (GG + (size_t)GG * HH) * sizeof(float), stream);
  hipMemsetAsync(zp, 0, (size_t)GG * ZKP * sizeof(float), stream);
  hipMemsetAsync(wpad, 0, (size_t)ZKP * HH * sizeof(float), stream);
  hipMemcpyAsync(wpad, fW1, (size_t)(HH + AA) * HH * sizeof(float),
                 hipMemcpyDeviceToDevice, stream);   // rows 288..319 stay zero

  deg_kernel  <<<(EE + T - 1) / T, T, 0, stream>>>(ei, dis, EE);
  dis_kernel  <<<(NN + T - 1) / T, T, 0, stream>>>(dis, NN);
  enorm_kernel<<<(EE + T - 1) / T, T, 0, stream>>>(ei, dis, enorm, EE);

  const int nhBlocks = (int)((NHf + T - 1) / T);
  const int edgeWaveBlocks = (EE + 7) / 8;      // 8 waves (edges) / 256-thread block
  const int statRows = 256;
  const int statBlocks = (NN + statRows - 1) / statRows;

  // ---- GCN layer 1 (GEMM fuses self-loop + bias init into buf1) ----
  gemm_wmma_bf16<<<gemmGrdN, gemmBlk, 0, stream>>>(x, W1, buf0, buf1, dis, b1, NN, FF, HH);
  scatter_kernel<<<edgeWaveBlocks, T, 0, stream>>>(buf0, ei, enorm, buf1, EE);
  hipMemsetAsync(mu, 0, 2 * HH * sizeof(float), stream);
  colpartial_kernel<<<statBlocks, HH, 0, stream>>>(buf1, mu, rstd, NN, statRows);
  colfinalize_kernel<<<1, HH, 0, stream>>>(mu, rstd, NN);
  bn_apply_kernel<<<nhBlocks, T, 0, stream>>>(buf1, mu, rstd, g1, be1, NN, HH);

  // ---- GCN layer 2 ----
  gemm_wmma_bf16<<<gemmGrdN, gemmBlk, 0, stream>>>(buf1, W2, buf0, buf1, dis, b2, NN, HH, HH);
  scatter_kernel<<<edgeWaveBlocks, T, 0, stream>>>(buf0, ei, enorm, buf1, EE);
  hipMemsetAsync(mu, 0, 2 * HH * sizeof(float), stream);
  colpartial_kernel<<<statBlocks, HH, 0, stream>>>(buf1, mu, rstd, NN, statRows);
  colfinalize_kernel<<<1, HH, 0, stream>>>(mu, rstd, NN);
  bn_apply_kernel<<<nhBlocks, T, 0, stream>>>(buf1, mu, rstd, g2, be2, NN, HH);

  // ---- GCN layer 3 (relu only) ----
  gemm_wmma_bf16<<<gemmGrdN, gemmBlk, 0, stream>>>(buf1, W3, buf0, buf1, dis, b3, NN, HH, HH);
  scatter_kernel<<<edgeWaveBlocks, T, 0, stream>>>(buf0, ei, enorm, buf1, EE);
  relu_kernel<<<nhBlocks, T, 0, stream>>>(buf1, NHf);

  // ---- global mean pool ----
  pool_count_kernel<<<(NN + T - 1) / T, T, 0, stream>>>(batch, counts);
  pool_sum_kernel<<<nhBlocks, T, 0, stream>>>(buf1, batch, pooled);
  pool_div_kernel<<<(GG * HH + T - 1) / T, T, 0, stream>>>(pooled, counts);

  // ---- head ----
  build_z_kernel<<<(GG * (HH + AA) + T - 1) / T, T, 0, stream>>>(pooled, actions, zp);

  gemm_wmma_bf16<<<dim3(1, HH / GEMM_BN), gemmBlk, 0, stream>>>(
      zp, wpad, z1, nullptr, nullptr, nullptr, GG, ZKP, HH);
  bias_relu_kernel<<<(GG * HH + T - 1) / T, T, 0, stream>>>(z1, fb1, GG, HH);
  hipMemsetAsync(mu, 0, 2 * HH * sizeof(float), stream);
  colpartial_kernel<<<1, HH, 0, stream>>>(z1, mu, rstd, GG, GG);
  colfinalize_kernel<<<1, HH, 0, stream>>>(mu, rstd, GG);
  bn_apply_kernel<<<(GG * HH + T - 1) / T, T, 0, stream>>>(z1, mu, rstd, g3, be3, GG, HH);

  gemm_wmma_bf16<<<dim3(1, (HH / 2) / GEMM_BN), gemmBlk, 0, stream>>>(
      z1, fW2, z2, nullptr, nullptr, nullptr, GG, HH, HH / 2);
  bias_relu_kernel<<<(GG * (HH / 2) + T - 1) / T, T, 0, stream>>>(z2, fb2, GG, HH / 2);

  head_final_kernel<<<GG, 128, 0, stream>>>(z2, fW3, fb3, out);
}
